// HypGRU_11278584119450
// MI455X (gfx1250) — matrix operand order repaired and checked by reading, not compiled, not run on hardware
//

#include <hip/hip_runtime.h>
#include <hip/hip_bf16.h>
#include <math.h>

// The Python harness owns main(), I/O, device malloc/memcpy, and timing.
// We implement kernel_launch() + kernels. All launches on `stream`.

typedef __bf16 v16bf __attribute__((ext_vector_type(16)));
typedef __bf16 v8bf  __attribute__((ext_vector_type(8)));
typedef float  v8f   __attribute__((ext_vector_type(8)));
typedef float  v4f   __attribute__((ext_vector_type(4)));

#define NHID     256
#define ROWS     32          // rows per workgroup (~144 KB LDS -> 2 WGs/WGP)
#define NWAVES   8
#define RPW      (ROWS / NWAVES)   // rows per wave in elementwise phases
#define KSTEPS   (NHID / 32)       // 8 k-stages per GEMM
#define MIN_NORM 1e-15f
#define PROJ_MAX 0.996f      // (1 - 4e-3)/sqrt(c), c = 1

// Scheduling fence: forbid the machine scheduler from sinking prefetch loads
// below the WMMAs of the current pipeline stage (keeps >=2 stages in flight).
#if __has_builtin(__builtin_amdgcn_sched_barrier)
#define SCHED_FENCE() __builtin_amdgcn_sched_barrier(0)
#else
#define SCHED_FENCE() ((void)0)
#endif

// ---------------- warp helpers (wave32) ----------------
__device__ __forceinline__ float wsum(float v) {
#pragma unroll
    for (int off = 16; off > 0; off >>= 1) v += __shfl_xor(v, off, 32);
    return v;
}
__device__ __forceinline__ float artanhf_(float x) {
    x = fminf(fmaxf(x, -1.0f + 1e-7f), 1.0f - 1e-7f);
    return 0.5f * (log1pf(x) - log1pf(-x));
}
__device__ __forceinline__ float sigmoidf_(float x) {
    return 1.0f / (1.0f + __expf(-x));
}

// mscale: v <- tanh(|v|/xn * artanh(xn)) * v/|v|   (c = 1)
__device__ __forceinline__ void mscale8(float v[8], float xn) {
    float s = 0.f;
#pragma unroll
    for (int u = 0; u < 8; ++u) s += v[u] * v[u];
    float mn = fmaxf(sqrtf(wsum(s)), MIN_NORM);
    float sc = tanhf(mn / xn * artanhf_(xn)) / mn;
#pragma unroll
    for (int u = 0; u < 8; ++u) v[u] *= sc;
}

// Mobius add: x <- mobius_add(x, y)   (c = 1)
__device__ __forceinline__ void madd8(float x[8], const float y[8]) {
    float sx = 0.f, sy = 0.f, sxy = 0.f;
#pragma unroll
    for (int u = 0; u < 8; ++u) {
        sx  += x[u] * x[u];
        sy  += y[u] * y[u];
        sxy += x[u] * y[u];
    }
    sx = wsum(sx); sy = wsum(sy); sxy = wsum(sxy);
    float ca  = 1.f + 2.f * sxy + sy;
    float cb  = 1.f - sx;
    float inv = 1.f / fmaxf(1.f + 2.f * sxy + sx * sy, MIN_NORM);
#pragma unroll
    for (int u = 0; u < 8; ++u) x[u] = (ca * x[u] + cb * y[u]) * inv;
}

// ---------------- WMMA fragment loaders ----------------
// A-matrix 16x32 bf16 layout: lane m (0..15) holds K = k0..k0+15 (contiguous,
// matching B); lanes 16..31 hold K = k0+16..k0+31.
// A lives in LDS as bf16, row-major with ld = NHID.
__device__ __forceinline__ v16bf load_fragA(const __bf16* A, int r0, int k0, int lane) {
    const int m = lane & 15, sel = lane >> 4;
    const __bf16* p = A + (size_t)(r0 + m) * NHID + k0 + sel * 16;
    return *(const v16bf*)(p);
}
// B-matrix 32x16 bf16: lane n (0..15) = output column; lanes 0-15 hold K=k0..k0+15,
// lanes 16-31 hold K=k0+16..k0+31 (contiguous per lane). Since we need W^T tile,
// lane n reads row (n0+n) of W: one contiguous 32B load from global (L2-resident).
__device__ __forceinline__ v16bf load_fragB(const __bf16* W, int n0, int k0, int lane) {
    const int n = lane & 15, sel = lane >> 4;
    return *(const v16bf*)(W + (size_t)(n0 + n) * NHID + k0 + sel * 16);
}

// ---------------- 32x256 GEMM: S = A @ W^T (fp32 accum to LDS) ----------------
// 8 waves: wave w -> row tile (w&1), col tiles (w>>1)*4 .. +3
// Software-pipelined, prefetch distance 2 (3-buffer rotation). A sched_barrier
// between the prefetch block and the WMMA block pins loads of stage s+2 above
// the WMMAs of stage s, so the allocator must keep the stages in distinct
// registers and waitcnts become partial instead of full drains.
__device__ __forceinline__ void gemm32(const __bf16* A, const __bf16* W, float* Sout, int tid) {
    const int lane = tid & 31;
    const int wave = tid >> 5;
    const int rt = wave & 1;
    const int cg = wave >> 1;
    v8f acc[4];
#pragma unroll
    for (int i = 0; i < 4; ++i)
#pragma unroll
        for (int j = 0; j < 8; ++j) acc[i][j] = 0.f;

    v16bf a[3];
    v16bf b[3][4];
    // prologue: stages 0 and 1
#pragma unroll
    for (int s = 0; s < 2; ++s) {
        a[s] = load_fragA(A, rt * 16, s * 32, lane);
#pragma unroll
        for (int ct = 0; ct < 4; ++ct)
            b[s][ct] = load_fragB(W, (cg * 4 + ct) * 16, s * 32, lane);
    }
#pragma unroll
    for (int s = 0; s < KSTEPS; ++s) {
        if (s + 2 < KSTEPS) {                 // prefetch stage s+2
            const int d = (s + 2) % 3;
            a[d] = load_fragA(A, rt * 16, (s + 2) * 32, lane);
#pragma unroll
            for (int ct = 0; ct < 4; ++ct)
                b[d][ct] = load_fragB(W, (cg * 4 + ct) * 16, (s + 2) * 32, lane);
        }
        SCHED_FENCE();                        // loads above may not sink below
        const int c = s % 3;
#pragma unroll
        for (int ct = 0; ct < 4; ++ct)
            acc[ct] = __builtin_amdgcn_wmma_f32_16x16x32_bf16(
                false, a[c], false, b[c][ct], (short)0, acc[ct], false, false);
        SCHED_FENCE();                        // keep stage boundaries intact
    }
    // C layout: VGPR r -> M = rt*16 + r + 8*sel, N = lane&15
    const int m0 = rt * 16 + (lane >> 4) * 8;
    const int n0 = cg * 64 + (lane & 15);
#pragma unroll
    for (int ct = 0; ct < 4; ++ct)
#pragma unroll
        for (int r = 0; r < 8; ++r)
            Sout[(m0 + r) * NHID + n0 + ct * 16] = acc[ct][r];
}

// one_rnn_transform elementwise tail for one row; returns norm of projected vec
__device__ __forceinline__ float transform_row(const float* Swh, const float* Sux,
                                               const float* brow, float na, float nb,
                                               int row, int lane, float p[8]) {
    float wh[8], ux[8], bb[8];
    const int base = row * NHID + lane;
#pragma unroll
    for (int u = 0; u < 8; ++u) {
        wh[u] = Swh[base + 32 * u];
        ux[u] = Sux[base + 32 * u];
        bb[u] = brow[lane + 32 * u];
    }
    mscale8(wh, na);        // mobius_matvec tail for W*h
    mscale8(ux, nb);        // mobius_matvec tail for U*x
    madd8(wh, ux);          // mobius_add
    madd8(wh, bb);          // mobius_add bias
    float s = 0.f;
#pragma unroll
    for (int u = 0; u < 8; ++u) s += wh[u] * wh[u];
    float nn = fmaxf(sqrtf(wsum(s)), MIN_NORM);
    float sc = (nn > PROJ_MAX) ? (PROJ_MAX / nn) : 1.0f;
#pragma unroll
    for (int u = 0; u < 8; ++u) p[u] = wh[u] * sc;
    return fmaxf(nn * sc, MIN_NORM);
}

// ---------------- weight fp32 -> bf16 pre-pass ----------------
__global__ void cvt_weights(const float* __restrict__ wih, const float* __restrict__ whh,
                            __bf16* __restrict__ out) {
    int i = blockIdx.x * blockDim.x + threadIdx.x;
    const int n = 3 * NHID * NHID;
    if (i < n) {
        out[i]     = (__bf16)wih[i];
        out[n + i] = (__bf16)whh[i];
    }
}

// ---------------- fused HypGRU kernel ----------------
__global__ void __launch_bounds__(256)
hypgru_fused(const float* __restrict__ x, const float* __restrict__ h,
             const __bf16* __restrict__ wih_bf, const __bf16* __restrict__ whh_bf,
             const float* __restrict__ bias, float* __restrict__ out) {
    extern __shared__ char smem[];
    float*  bufA = (float*)smem;                    // 32 KB  GEMM stage (W*h side)
    float*  bufB = bufA + ROWS * NHID;              // 32 KB  GEMM stage (U*x side)
    float*  bufZ = bufB + ROWS * NHID;              // 32 KB  z gate
    __bf16* xbf  = (__bf16*)(bufZ + ROWS * NHID);   // 16 KB  x tile (bf16 A)
    __bf16* hbf  = xbf + ROWS * NHID;               // 16 KB  h tile (bf16 A)
    __bf16* rhbf = hbf + ROWS * NHID;               // 16 KB  rh     (bf16 A)
    float*  hn   = (float*)(rhbf + ROWS * NHID);    // per-row norms
    float*  xn   = hn + ROWS;
    float*  rhn  = xn + ROWS;

    const int tid  = threadIdx.x;
    const int lane = tid & 31;
    const int wave = tid >> 5;
    const int row0 = blockIdx.x * ROWS;
    const float* xblk = x + (size_t)row0 * NHID;
    const float* hblk = h + (size_t)row0 * NHID;

    // phase 0: per-row norms of h and x; convert tiles to bf16 in LDS (read once)
#pragma unroll
    for (int i = 0; i < RPW; ++i) {
        int row = wave * RPW + i;
        float sh = 0.f, sx = 0.f;
#pragma unroll
        for (int u = 0; u < 8; ++u) {
            int idx = row * NHID + lane + 32 * u;
            float hv = hblk[idx];
            float xv = xblk[idx];
            hbf[idx] = (__bf16)hv;
            xbf[idx] = (__bf16)xv;
            sh += hv * hv;
            sx += xv * xv;
        }
        sh = wsum(sh); sx = wsum(sx);
        if (lane == 0) {
            hn[row] = fmaxf(sqrtf(sh), MIN_NORM);
            xn[row] = fmaxf(sqrtf(sx), MIN_NORM);
        }
    }
    __syncthreads();

    const __bf16* W_ir = wih_bf;
    const __bf16* W_ih = wih_bf + 256 * NHID;
    const __bf16* W_iz = wih_bf + 512 * NHID;
    const __bf16* W_hr = whh_bf;
    const __bf16* W_hh = whh_bf + 256 * NHID;
    const __bf16* W_hz = whh_bf + 512 * NHID;
    const float* b_r = bias;
    const float* b_h = bias + NHID;
    const float* b_z = bias + 2 * NHID;

    // ---- z gate ----
    gemm32(hbf, W_hz, bufA, tid);
    gemm32(xbf, W_iz, bufB, tid);
    __syncthreads();
#pragma unroll
    for (int i = 0; i < RPW; ++i) {
        int row = wave * RPW + i;
        float p[8];
        float pn  = transform_row(bufA, bufB, b_z, hn[row], xn[row], row, lane, p);
        float lsc = artanhf_(pn) / pn;    // logmap0 scale
#pragma unroll
        for (int u = 0; u < 8; ++u)
            bufZ[row * NHID + lane + 32 * u] = sigmoidf_(lsc * p[u]);
    }
    __syncthreads();

    // ---- r gate -> rh ----
    gemm32(hbf, W_hr, bufA, tid);
    gemm32(xbf, W_ir, bufB, tid);
    __syncthreads();
#pragma unroll
    for (int i = 0; i < RPW; ++i) {
        int row = wave * RPW + i;
        float p[8];
        float pn  = transform_row(bufA, bufB, b_r, hn[row], xn[row], row, lane, p);
        float lsc = artanhf_(pn) / pn;
        float rh[8];
#pragma unroll
        for (int u = 0; u < 8; ++u) {
            float rg = sigmoidf_(lsc * p[u]);
            rh[u] = rg * hblk[row * NHID + lane + 32 * u];
        }
        // mobius_pointwise_mul tail: mscale(rh, |h|), record resulting norm
        float s = 0.f;
#pragma unroll
        for (int u = 0; u < 8; ++u) s += rh[u] * rh[u];
        float mn  = fmaxf(sqrtf(wsum(s)), MIN_NORM);
        float hnr = hn[row];
        float tv  = tanhf(mn / hnr * artanhf_(hnr));
        float sc2 = tv / mn;
#pragma unroll
        for (int u = 0; u < 8; ++u)
            rhbf[row * NHID + lane + 32 * u] = (__bf16)(rh[u] * sc2);
        if (lane == 0) rhn[row] = fmaxf(tv, MIN_NORM);
    }
    __syncthreads();

    // ---- h_tilde ----
    gemm32(rhbf, W_hh, bufA, tid);
    gemm32(xbf,  W_ih, bufB, tid);
    __syncthreads();
#pragma unroll
    for (int i = 0; i < RPW; ++i) {
        int row = wave * RPW + i;
        float p[8];
        (void)transform_row(bufA, bufB, b_h, rhn[row], xn[row], row, lane, p);
        float ht[8], hv[8], dl[8];
#pragma unroll
        for (int u = 0; u < 8; ++u) {
            ht[u] = tanhf(p[u]);                          // h_tilde
            hv[u] = hblk[row * NHID + lane + 32 * u];
            dl[u] = -hv[u];
        }
        madd8(dl, ht);                                    // delta = madd(-h, h_tilde)
        float s = 0.f;
#pragma unroll
        for (int u = 0; u < 8; ++u) s += dl[u] * dl[u];
        float dn = fmaxf(sqrtf(wsum(s)), MIN_NORM);
        float zd[8];
#pragma unroll
        for (int u = 0; u < 8; ++u)
            zd[u] = bufZ[row * NHID + lane + 32 * u] * dl[u];
        mscale8(zd, dn);                                  // zdelta
        madd8(hv, zd);                                    // h_out = madd(h, zdelta)
#pragma unroll
        for (int u = 0; u < 8; ++u)
            out[(size_t)(row0 + row) * NHID + lane + 32 * u] = hv[u];
    }
}

extern "C" void kernel_launch(void* const* d_in, const int* in_sizes, int n_in,
                              void* d_out, int out_size, void* d_ws, size_t ws_size,
                              hipStream_t stream) {
    (void)n_in; (void)out_size; (void)ws_size;
    const float* x    = (const float*)d_in[0];   // hyperx (65536,256)
    const float* h    = (const float*)d_in[1];   // hyperh (65536,256)
    const float* wih  = (const float*)d_in[2];   // (768,256)
    const float* whh  = (const float*)d_in[3];   // (768,256)
    const float* bias = (const float*)d_in[4];   // (3,256)
    float* out = (float*)d_out;

    __bf16* wbf = (__bf16*)d_ws;                 // 2 * 768*256 bf16 = 768 KB
    const int nw = 3 * NHID * NHID;
    cvt_weights<<<(nw + 255) / 256, 256, 0, stream>>>(wih, whh, wbf);

    const int nrows = in_sizes[0] / NHID;        // 65536
    const int grid  = nrows / ROWS;              // 2048 workgroups
    const size_t smem_bytes = (size_t)(3 * ROWS * NHID) * sizeof(float)     // bufA/B/Z
                            + (size_t)(3 * ROWS * NHID) * sizeof(__bf16)    // xbf/hbf/rhbf
                            + (size_t)(3 * ROWS) * sizeof(float);           // norms  (~144 KB)
    (void)hipFuncSetAttribute((const void*)hypgru_fused,
                              hipFuncAttributeMaxDynamicSharedMemorySize,
                              (int)smem_bytes);
    hypgru_fused<<<grid, 256, smem_bytes, stream>>>(x, h, wbf, wbf + nw, bias, out);
}
